// Anchor3DHead_50148038148213
// MI455X (gfx1250) — matrix operand order repaired and checked by reading, not compiled, not run on hardware
//
#include <hip/hip_runtime.h>

#define C_IN   384
#define HGT    248
#define WID    216
#define NBATCH 4
#define PIX    (HGT * WID)        // 53568
#define NPIX   (NBATCH * PIX)     // 214272
#define MTILE  128
#define NBLK   (NPIX / MTILE)     // 1674 (exact)
#define KCHUNK 64
#define NKC    (C_IN / KCHUNK)    // 6
#define ASTR   72                 // LDS row stride in halves (16B-aligned rows)
#define WSTR   72
#define OROWS  80                 // 72 real outputs padded to 5 n-tiles

typedef __attribute__((ext_vector_type(8)))  unsigned short u16x8;
typedef __attribute__((ext_vector_type(16))) unsigned short u16x16;
typedef __attribute__((ext_vector_type(16))) __bf16         v16bf;
typedef __attribute__((ext_vector_type(8)))  float          v8f;

// round-to-nearest-even f32 -> bf16
static __device__ __forceinline__ unsigned short f2bf(float f) {
    unsigned int u = __builtin_bit_cast(unsigned int, f);
    u += 0x7FFFu + ((u >> 16) & 1u);
    return (unsigned short)(u >> 16);
}

// assemble a 16-half fragment from two 16-byte LDS chunks
static __device__ __forceinline__ v16bf frag16(const unsigned short* p0,
                                               const unsigned short* p1) {
    u16x8 lo = *(const u16x8*)p0;
    u16x8 hi = *(const u16x8*)p1;
    u16x16 full = __builtin_shufflevector(lo, hi,
        0,1,2,3,4,5,6,7, 8,9,10,11,12,13,14,15);
    return __builtin_bit_cast(v16bf, full);
}

__global__ __launch_bounds__(256)
void anchor3dhead_wmma_bf16(const float* __restrict__ x,
                            const float* __restrict__ w_cls, const float* __restrict__ b_cls,
                            const float* __restrict__ w_reg, const float* __restrict__ b_reg,
                            const float* __restrict__ w_dir, const float* __restrict__ b_dir,
                            float* __restrict__ out)
{
    __shared__ unsigned short Abuf[MTILE * ASTR];   // 18432 B
    __shared__ unsigned short Wbuf[OROWS * WSTR];   // 11520 B

    const int tid  = threadIdx.x;
    const int wave = tid >> 5;
    const int lane = tid & 31;
    const int col  = lane & 15;       // A row (pixel) / B column (output ch)
    const int half = lane >> 4;       // lane half selects K sub-group
    const int tile_base = blockIdx.x * MTILE;

    // --- X staging addressing (thread-invariant across k-chunks) ---
    const int p_local = tid & 127;
    const int c0      = tid >> 7;                 // 0 or 1
    {
        // computed below, kept per-thread
    }
    const unsigned int n_st = (unsigned int)(tile_base + p_local);
    const unsigned int bs   = n_st / PIX;
    const unsigned int ps   = n_st - bs * PIX;
    const float* xb = x + (size_t)bs * C_IN * PIX + ps;

    // --- W staging addressing ---
    const int wc  = tid & 63;
    const int wr0 = tid >> 6;                     // 0..3

    v8f acc[5];
    v8f zero = {};
    #pragma unroll
    for (int i = 0; i < 5; ++i) acc[i] = zero;

    const unsigned short* arow  = &Abuf[(wave * 16 + col) * ASTR];
    const unsigned short* wbase = &Wbuf[col * WSTR];

    for (int kc = 0; kc < NKC; ++kc) {
        __syncthreads();
        // stage X: 128 pixels x 64 channels, f32 -> bf16
        #pragma unroll 8
        for (int c = 0; c < KCHUNK; c += 2) {
            int cl = c + c0;
            float v = xb[(size_t)(kc * KCHUNK + cl) * PIX];
            Abuf[p_local * ASTR + cl] = f2bf(v);
        }
        // stage W: 80 rows x 64 channels (rows >= 72 are zero padding)
        #pragma unroll 4
        for (int r = wr0; r < OROWS; r += 4) {
            float v = 0.f;
            int k = kc * KCHUNK + wc;
            if (r < 18)      v = w_cls[r * C_IN + k];
            else if (r < 60) v = w_reg[(r - 18) * C_IN + k];
            else if (r < 72) v = w_dir[(r - 60) * C_IN + k];
            Wbuf[r * WSTR + wc] = f2bf(v);
        }
        __syncthreads();

        #pragma unroll
        for (int kk = 0; kk < KCHUNK; kk += 32) {
            // A (16x32 bf16): halves 0-7 = K kk+8*half.., halves 8-15 = K kk+16+8*half..
            v16bf afrag = frag16(arow + kk + half * 8,
                                 arow + kk + 16 + half * 8);
            #pragma unroll
            for (int nt = 0; nt < 5; ++nt) {
                // B (32x16 bf16): halves 0-7 = K kk+8*half.., halves 8-15 = K kk+16+8*half..
                const unsigned short* brow = wbase + nt * 16 * WSTR + kk;
                v16bf bfrag = frag16(brow + half * 8, brow + 16 + half * 8);
                acc[nt] = __builtin_amdgcn_wmma_f32_16x16x32_bf16(
                    false, afrag, false, bfrag, (short)0, acc[nt], false, false);
            }
        }
    }

    // --- epilogue: each lane owns 8 consecutive pixels of one output channel ---
    const unsigned int n0 = (unsigned int)(tile_base + wave * 16 + half * 8);
    const unsigned int b  = n0 / PIX;       // PIX % 8 == 0 -> 8-run never crosses batch
    const unsigned int p0 = n0 - b * PIX;

    #pragma unroll
    for (int nt = 0; nt < 5; ++nt) {
        int o = nt * 16 + col;
        if (o >= 72) continue;              // padding columns (nt==4, col>=8)
        float bias;
        float* dst;
        if (o < 18) {
            bias = b_cls[o];
            dst  = out + ((size_t)b * 18 + o) * PIX + p0;
        } else if (o < 60) {
            bias = b_reg[o - 18];
            dst  = out + (size_t)NBATCH * 18 * PIX
                       + ((size_t)b * 42 + (o - 18)) * PIX + p0;
        } else {
            bias = b_dir[o - 60];
            dst  = out + (size_t)NBATCH * 60 * PIX
                       + ((size_t)b * 12 + (o - 60)) * PIX + p0;
        }
        v8f a = acc[nt];
        float4 lo = make_float4(a[0] + bias, a[1] + bias, a[2] + bias, a[3] + bias);
        float4 hi = make_float4(a[4] + bias, a[5] + bias, a[6] + bias, a[7] + bias);
        *(float4*)dst       = lo;
        *((float4*)dst + 1) = hi;
    }
}

extern "C" void kernel_launch(void* const* d_in, const int* in_sizes, int n_in,
                              void* d_out, int out_size, void* d_ws, size_t ws_size,
                              hipStream_t stream) {
    (void)in_sizes; (void)n_in; (void)out_size; (void)d_ws; (void)ws_size;
    const float* x     = (const float*)d_in[0];
    const float* w_cls = (const float*)d_in[1];
    const float* b_cls = (const float*)d_in[2];
    const float* w_reg = (const float*)d_in[3];
    const float* b_reg = (const float*)d_in[4];
    const float* w_dir = (const float*)d_in[5];
    const float* b_dir = (const float*)d_in[6];
    float* out = (float*)d_out;

    anchor3dhead_wmma_bf16<<<dim3(NBLK), dim3(256), 0, stream>>>(
        x, w_cls, b_cls, w_reg, b_reg, w_dir, b_dir, out);
}
